// DistributedSparseAttention_43860206027370
// MI455X (gfx1250) — matrix operand order repaired and checked by reading, not compiled, not run on hardware
//
#include <hip/hip_runtime.h>
#include <hip/hip_bf16.h>

// ---------------------------------------------------------------------------
// DistributedSparseAttention for MI455X (gfx1250), wave32 + WMMA f16.
// ---------------------------------------------------------------------------

typedef __attribute__((ext_vector_type(16))) _Float16 v16h;
typedef __attribute__((ext_vector_type(8)))  _Float16 v8h;
typedef __attribute__((ext_vector_type(8)))  float    v8f;

#define CB 4
#define CS 2048
#define CD 1024
#define CH 16
#define CHD 64
#define CU 38
#define CM (CB*CS)        // 8192
#define CBH (CB*CH)       // 64
#define SCALE 0.125f      // 1/sqrt(64)

__device__ __forceinline__ v16h cat8(v8h lo, v8h hi) {
    return __builtin_shufflevector(lo, hi, 0,1,2,3,4,5,6,7,8,9,10,11,12,13,14,15);
}

__device__ __forceinline__ v8f wmma_f16(v16h a, v16h b, v8f c) {
    return __builtin_amdgcn_wmma_f32_16x16x32_f16(false, a, false, b, (short)0, c, false, false);
}

// Load A fragment (16x32, f16 layout) from f16 row-major source, row stride ld
__device__ __forceinline__ v16h load_a_f16(const _Float16* A, int mbase, int lane, int k0, int ld) {
    int r = lane & 15, g = lane >> 4;
    const _Float16* p = A + (size_t)(mbase + r) * ld + k0 + 8 * g;
    v8h lo = *(const v8h*)p;
    v8h hi = *(const v8h*)(p + 16);
    return cat8(lo, hi);
}

// Load A fragment from f32 row-major source, converting to f16
__device__ __forceinline__ v16h load_a_f32(const float* A, int mbase, int lane, int k0, int ld) {
    int r = lane & 15, g = lane >> 4;
    const float* p = A + (size_t)(mbase + r) * ld + k0 + 8 * g;
    v8f f0 = *(const v8f*)p;
    v8f f1 = *(const v8f*)(p + 16);
    v8h lo = __builtin_convertvector(f0, v8h);
    v8h hi = __builtin_convertvector(f1, v8h);
    return cat8(lo, hi);
}

// Load B fragment (32x16): B[k][n] = W[nbase+n][k]; lane n -> halves k0+8g..+7 and k0+16+8g..+7
__device__ __forceinline__ v16h load_b(const _Float16* W, int nbase, int lane, int k0, int ld) {
    int n = lane & 15, g = lane >> 4;
    const _Float16* p = W + (size_t)(nbase + n) * ld + k0 + 8 * g;
    return cat8(*(const v8h*)p, *(const v8h*)(p + 16));
}

// ---------------------------------------------------------------------------
// f32 -> f16 conversion
// ---------------------------------------------------------------------------
__global__ void cvt_f32_f16(const float* __restrict__ src, _Float16* __restrict__ dst, int n) {
    int i = blockIdx.x * 256 + threadIdx.x;
    if (i < n) dst[i] = (_Float16)src[i];
}

// ---------------------------------------------------------------------------
// GEMM: out(M,1024) = A(M,1024) * W(1024,1024)^T + bias
// MODE 0: f32 out, (B,S,D) layout.  MODE 1: f16 out, (B,H,S,HD).  MODE 2: f32, (B,H,S,HD)
// block = 128 threads (4 waves), each wave does a 16x64 strip; grid (N/64, M/64)
// ---------------------------------------------------------------------------
template<int MODE, bool AF32>
__global__ __launch_bounds__(128)
void gemm_xwt(const void* __restrict__ A_, const _Float16* __restrict__ W,
              const float* __restrict__ bias, void* __restrict__ out_) {
    const int K = CD;
    int wave = threadIdx.x >> 5, lane = threadIdx.x & 31;
    int g = lane >> 4;
    int mbase = blockIdx.y * 64 + wave * 16;
    int nbase = blockIdx.x * 64;

    v8f c[4];
    #pragma unroll
    for (int j = 0; j < 4; ++j) c[j] = (v8f){0,0,0,0,0,0,0,0};

    for (int k0 = 0; k0 < K; k0 += 32) {
        v16h a = AF32 ? load_a_f32((const float*)A_, mbase, lane, k0, K)
                      : load_a_f16((const _Float16*)A_, mbase, lane, k0, K);
        v16h b0 = load_b(W, nbase +  0, lane, k0, K);
        v16h b1 = load_b(W, nbase + 16, lane, k0, K);
        v16h b2 = load_b(W, nbase + 32, lane, k0, K);
        v16h b3 = load_b(W, nbase + 48, lane, k0, K);
        c[0] = wmma_f16(a, b0, c[0]);
        c[1] = wmma_f16(a, b1, c[1]);
        c[2] = wmma_f16(a, b2, c[2]);
        c[3] = wmma_f16(a, b3, c[3]);
    }

    int col0 = nbase + (lane & 15);
    #pragma unroll
    for (int j = 0; j < 4; ++j) {
        int col = col0 + 16 * j;
        float bb = bias[col];
        #pragma unroll
        for (int rr = 0; rr < 8; ++rr) {
            int row = mbase + rr + 8 * g;
            float val = c[j][rr] + bb;
            if (MODE == 0) {
                ((float*)out_)[(size_t)row * CD + col] = val;
            } else {
                int b = row >> 11;          // / CS
                int s = row & (CS - 1);
                int h = col >> 6;           // / CHD
                int hd = col & (CHD - 1);
                size_t idx = (((size_t)(b * CH + h) * CS + s) * CHD) + hd;
                if (MODE == 1) ((_Float16*)out_)[idx] = (_Float16)val;
                else           ((float*)out_)[idx] = val;
            }
        }
    }
}

// ---------------------------------------------------------------------------
// Score scan: per (b,h), importance[q] = max_k(score) - mean_k(score)
// One wave per 16-query strip; loops over all 128 key tiles, WMMA 16x16x32 x2 (HD=64)
// grid (S/16/8, B*H), block 256 (8 waves)
// ---------------------------------------------------------------------------
__global__ __launch_bounds__(256)
void scores_importance(const _Float16* __restrict__ Qh, const _Float16* __restrict__ Kh,
                       float* __restrict__ imp) {
    int wave = threadIdx.x >> 5, lane = threadIdx.x & 31;
    int g = lane >> 4;
    int bh = blockIdx.y;
    int mbase = (blockIdx.x * 8 + wave) * 16;
    const _Float16* Qb = Qh + (size_t)bh * CS * CHD;
    const _Float16* Kb = Kh + (size_t)bh * CS * CHD;

    v16h A0 = load_a_f16(Qb, mbase, lane, 0, CHD);
    v16h A1 = load_a_f16(Qb, mbase, lane, 32, CHD);

    v8f pmax = (v8f){-3e38f,-3e38f,-3e38f,-3e38f,-3e38f,-3e38f,-3e38f,-3e38f};
    v8f psum = (v8f){0,0,0,0,0,0,0,0};

    for (int nt = 0; nt < CS; nt += 16) {
        v16h B0 = load_b(Kb, nt, lane, 0, CHD);
        v16h B1 = load_b(Kb, nt, lane, 32, CHD);
        v8f c = (v8f){0,0,0,0,0,0,0,0};
        c = wmma_f16(A0, B0, c);
        c = wmma_f16(A1, B1, c);
        #pragma unroll
        for (int rr = 0; rr < 8; ++rr) {
            float s = c[rr] * SCALE;
            psum[rr] += s;
            pmax[rr] = fmaxf(pmax[rr], s);
        }
    }
    // reduce across the 16 lanes holding each row's 16 columns
    #pragma unroll
    for (int rr = 0; rr < 8; ++rr) {
        float m = pmax[rr], su = psum[rr];
        #pragma unroll
        for (int off = 8; off > 0; off >>= 1) {
            m = fmaxf(m, __shfl_xor(m, off, 16));
            su += __shfl_xor(su, off, 16);
        }
        if ((lane & 15) == 0)
            imp[(size_t)bh * CS + mbase + rr + 8 * g] = m - su * (1.0f / CS);
    }
}

// ---------------------------------------------------------------------------
// Iterative top-U argmax per (b,h). grid = B*H, block 256.
// ---------------------------------------------------------------------------
__global__ __launch_bounds__(256)
void topk_kernel(const float* __restrict__ imp, int* __restrict__ sel) {
    __shared__ float vals[CS];
    __shared__ float rv[256];
    __shared__ int   ri[256];
    int bh = blockIdx.x, t = threadIdx.x;
    for (int i = t; i < CS; i += 256) vals[i] = imp[(size_t)bh * CS + i];
    __syncthreads();
    for (int u = 0; u < CU; ++u) {
        float bvv = -3e38f; int bi = 1 << 30;
        for (int i = t; i < CS; i += 256) {
            float v = vals[i];
            if (v > bvv) { bvv = v; bi = i; }
        }
        rv[t] = bvv; ri[t] = bi;
        __syncthreads();
        for (int off = 128; off > 0; off >>= 1) {
            if (t < off) {
                float v2 = rv[t + off]; int i2 = ri[t + off];
                if (v2 > rv[t] || (v2 == rv[t] && i2 < ri[t])) { rv[t] = v2; ri[t] = i2; }
            }
            __syncthreads();
        }
        if (t == 0) { sel[bh * CU + u] = ri[0]; vals[ri[0]] = -3e38f; }
        __syncthreads();
    }
}

// ---------------------------------------------------------------------------
// Mean of V over keys: vm[b,h,hd]. grid = B*H, block 64.
// ---------------------------------------------------------------------------
__global__ __launch_bounds__(64)
void vmean_kernel(const float* __restrict__ V, float* __restrict__ vm) {
    int bh = blockIdx.x, hd = threadIdx.x;
    const float* p = V + (size_t)bh * CS * CHD + hd;
    float s = 0.f;
    for (int k = 0; k < CS; ++k) s += p[(size_t)k * CHD];
    vm[bh * CHD + hd] = s * (1.0f / CS);
}

// ---------------------------------------------------------------------------
// Fill attention output with per-head default (mean of V), f16 (B,S,D).
// ---------------------------------------------------------------------------
__global__ __launch_bounds__(256)
void compose_default(const float* __restrict__ vm, _Float16* __restrict__ attn) {
    size_t i = (size_t)blockIdx.x * 256 + threadIdx.x;   // over B*S*D
    int d = (int)(i & (CD - 1));
    size_t bs = i >> 10;
    int b = (int)(bs >> 11);
    int h = d >> 6, hd = d & (CHD - 1);
    attn[i] = (_Float16)vm[(b * CH + h) * CHD + hd];
}

// ---------------------------------------------------------------------------
// Softmax attention for selected queries; overwrites their rows in attn.
// grid (U, B*H), block 64 (one thread per head-dim in final pass).
// ---------------------------------------------------------------------------
__global__ __launch_bounds__(64)
void sel_attn(const _Float16* __restrict__ Qh, const _Float16* __restrict__ Kh,
              const float* __restrict__ V, const int* __restrict__ sel,
              _Float16* __restrict__ attn) {
    __shared__ float q[CHD];
    __shared__ float sv[CS];
    __shared__ float red[64];
    int bh = blockIdx.y, u = blockIdx.x, t = threadIdx.x;
    int b = bh >> 4, h = bh & 15;
    int sq = sel[bh * CU + u];

    q[t] = (float)Qh[((size_t)bh * CS + sq) * CHD + t];
    __syncthreads();

    const _Float16* Kb = Kh + (size_t)bh * CS * CHD;
    float lmax = -3e38f;
    for (int k = t; k < CS; k += 64) {
        const _Float16* kp = Kb + (size_t)k * CHD;
        float dot = 0.f;
        #pragma unroll 8
        for (int i = 0; i < CHD; ++i) dot += q[i] * (float)kp[i];
        dot *= SCALE;
        sv[k] = dot;
        lmax = fmaxf(lmax, dot);
    }
    red[t] = lmax; __syncthreads();
    for (int off = 32; off > 0; off >>= 1) {
        if (t < off) red[t] = fmaxf(red[t], red[t + off]);
        __syncthreads();
    }
    float mx = red[0];
    __syncthreads();

    float lsum = 0.f;
    for (int k = t; k < CS; k += 64) {
        float e = __expf(sv[k] - mx);
        sv[k] = e;
        lsum += e;
    }
    red[t] = lsum; __syncthreads();
    for (int off = 32; off > 0; off >>= 1) {
        if (t < off) red[t] += red[t + off];
        __syncthreads();
    }
    float inv = 1.0f / (red[0] + 1e-8f);
    __syncthreads();

    const float* Vb = V + (size_t)bh * CS * CHD;
    float acc = 0.f;
    for (int k = 0; k < CS; ++k) acc += sv[k] * Vb[(size_t)k * CHD + t];

    attn[((size_t)b * CS + sq) * CD + h * CHD + t] = (_Float16)(acc * inv);
}

// ---------------------------------------------------------------------------
// Host launcher
// ---------------------------------------------------------------------------
extern "C" void kernel_launch(void* const* d_in, const int* in_sizes, int n_in,
                              void* d_out, int out_size, void* d_ws, size_t ws_size,
                              hipStream_t stream) {
    const float* q_in = (const float*)d_in[0];
    const float* k_in = (const float*)d_in[1];
    const float* v_in = (const float*)d_in[2];
    const float* Wq   = (const float*)d_in[3];
    const float* bq   = (const float*)d_in[4];
    const float* Wk   = (const float*)d_in[5];
    const float* bk   = (const float*)d_in[6];
    const float* Wv   = (const float*)d_in[7];
    const float* bv   = (const float*)d_in[8];
    const float* Wo   = (const float*)d_in[9];
    const float* bo   = (const float*)d_in[10];

    char* ws = (char*)d_ws;
    const size_t WSZ   = (size_t)CD * CD;              // 1M elems per weight
    _Float16* Wqh  = (_Float16*)(ws);
    _Float16* Wkh  = Wqh + WSZ;
    _Float16* Wvh  = Wkh + WSZ;
    _Float16* Woh  = Wvh + WSZ;
    size_t off = 4 * WSZ * sizeof(_Float16);           // 8 MB
    _Float16* Qh   = (_Float16*)(ws + off);  off += (size_t)CBH * CS * CHD * 2;  // 16 MB
    _Float16* Kh   = (_Float16*)(ws + off);  off += (size_t)CBH * CS * CHD * 2;  // 16 MB
    float*    Vf   = (float*)   (ws + off);  off += (size_t)CBH * CS * CHD * 4;  // 32 MB
    _Float16* attn = (_Float16*)(ws + off);  off += (size_t)CM * CD * 2;         // 16 MB
    float*    imp  = (float*)   (ws + off);  off += (size_t)CBH * CS * 4;        // 512 KB
    float*    vm   = (float*)   (ws + off);  off += (size_t)CBH * CHD * 4;
    int*      sel  = (int*)     (ws + off);

    // 1. convert weights to f16
    int wN = CD * CD;
    cvt_f32_f16<<<(wN + 255) / 256, 256, 0, stream>>>(Wq, Wqh, wN);
    cvt_f32_f16<<<(wN + 255) / 256, 256, 0, stream>>>(Wk, Wkh, wN);
    cvt_f32_f16<<<(wN + 255) / 256, 256, 0, stream>>>(Wv, Wvh, wN);
    cvt_f32_f16<<<(wN + 255) / 256, 256, 0, stream>>>(Wo, Woh, wN);

    // 2-4. Q/K/V projections (WMMA)
    dim3 ggrid(CD / 64, CM / 64);   // (16,128)
    gemm_xwt<1, true><<<ggrid, 128, 0, stream>>>((const void*)q_in, Wqh, bq, (void*)Qh);
    gemm_xwt<1, true><<<ggrid, 128, 0, stream>>>((const void*)k_in, Wkh, bk, (void*)Kh);
    gemm_xwt<2, true><<<ggrid, 128, 0, stream>>>((const void*)v_in, Wvh, bv, (void*)Vf);

    // 5. fused score scan -> importance (WMMA)
    scores_importance<<<dim3(CS / 16 / 8, CBH), 256, 0, stream>>>(Qh, Kh, imp);

    // 6. top-U selection
    topk_kernel<<<CBH, 256, 0, stream>>>(imp, sel);

    // 7. V mean
    vmean_kernel<<<CBH, 64, 0, stream>>>(Vf, vm);

    // 8. default fill
    compose_default<<<(int)(((size_t)CM * CD) / 256), 256, 0, stream>>>(vm, attn);

    // 9. selected-query attention (overwrites selected rows)
    sel_attn<<<dim3(CU, CBH), 64, 0, stream>>>(Qh, Kh, Vf, sel, attn);

    // 10. output projection (WMMA) -> d_out f32
    gemm_xwt<0, false><<<ggrid, 128, 0, stream>>>((const void*)attn, Woh, bo, d_out);
}
